// GalModel_8005819040480
// MI455X (gfx1250) — matrix operand order repaired and checked by reading, not compiled, not run on hardware
//
#include <hip/hip_runtime.h>
#include <hip/hip_bf16.h>
#include <math.h>

// ---------------------------------------------------------------------------
// Sizes (match reference)
// ---------------------------------------------------------------------------
#define NN   50000
#define FF   500
#define EE   1600000
#define PEE  400000
#define CC   3
#define H1C  128
#define H2C  64

#define CDIV(a,b) (((a)+(b)-1)/(b))

typedef __attribute__((ext_vector_type(16))) _Float16 v16h;
typedef __attribute__((ext_vector_type(8)))  float    v8f;

// ---------------------------------------------------------------------------
// Utility: fill
// ---------------------------------------------------------------------------
__global__ void k_fill(float* __restrict__ p, long long n, float v) {
    long long i = (long long)blockIdx.x * blockDim.x + threadIdx.x;
    if (i < n) p[i] = v;
}

// ---------------------------------------------------------------------------
// Weff = glove @ W1   [500,500]@[500,128] -> [500,128]  (folds glove into W1)
// ---------------------------------------------------------------------------
__global__ void k_weff(const float* __restrict__ glove, const float* __restrict__ W1,
                       float* __restrict__ Weff) {
    int t = blockIdx.x * blockDim.x + threadIdx.x;
    if (t >= FF * H1C) return;
    int r = t / H1C, c = t % H1C;
    float acc = 0.f;
    for (int k = 0; k < FF; ++k)
        acc = fmaf(glove[(size_t)r * FF + k], W1[(size_t)k * H1C + c], acc);
    Weff[t] = acc;
}

// ---------------------------------------------------------------------------
// GCN normalization: deg (self-loops pre-seeded to 1.0), dinv, per-edge norm
// ---------------------------------------------------------------------------
__global__ void k_deg(const int* __restrict__ dst, float* __restrict__ deg) {
    int e = blockIdx.x * blockDim.x + threadIdx.x;
    if (e < EE) atomicAdd(&deg[dst[e]], 1.0f);
}

__global__ void k_dinv(const float* __restrict__ deg, float* __restrict__ dinv) {
    int i = blockIdx.x * blockDim.x + threadIdx.x;
    if (i < NN) {
        float d = deg[i];
        dinv[i] = d > 0.f ? rsqrtf(d) : 0.f;
    }
}

__global__ void k_norm(const int* __restrict__ src, const int* __restrict__ dst,
                       const float* __restrict__ dinv, float* __restrict__ nrm) {
    int e = blockIdx.x * blockDim.x + threadIdx.x;
    if (e < EE) nrm[e] = dinv[src[e]] * dinv[dst[e]];
}

// ---------------------------------------------------------------------------
// Pack fp32 B[K,Nt*16] into per-lane WMMA f16 fragments, zero-padded in K.
// Fragment layout (CDNA5 ISA 7.12.2, B 32x16 16-bit):
//   lane = h*16 + l -> column N = tile*16 + l ; element j -> K = kchunk*32 + (j<8 ? h*8+j : 16+h*8+(j-8))
// Bp[(((t*kc + kchunk)*32 + lane)*16 + j]
// ---------------------------------------------------------------------------
__global__ void k_packB(const float* __restrict__ B, int ldb, int K, int ntiles,
                        _Float16* __restrict__ Bp) {
    const int kc = (K + 31) >> 5;
    const long long total = (long long)ntiles * kc * 32 * 16;
    long long tid = (long long)blockIdx.x * blockDim.x + threadIdx.x;
    if (tid >= total) return;
    const int j    = (int)(tid & 15);
    const int lane = (int)((tid >> 4) & 31);
    const long long rest = tid >> 9;
    const int kchunk = (int)(rest % kc);
    const int t      = (int)(rest / kc);
    const int h = lane >> 4, l = lane & 15;
    const int k   = (kchunk << 5) + (j < 8 ? h * 8 + j : 16 + h * 8 + (j - 8));
    const int col = t * 16 + l;
    float f = (k < K) ? B[(size_t)k * ldb + col] : 0.f;
    Bp[tid] = (_Float16)f;
}

// ---------------------------------------------------------------------------
// WMMA GEMM: C[M, NT*16] = A[M,K] * B[K, NT*16], fp32 A/C, f16 multiply.
// One wave32 per 16-row block; A fragment reused across NT column tiles.
// M multiple of 16 (50000 = 3125*16). B pre-packed by k_packB.
// A 16x32 f16 fragment: lane h*16+l -> row M=l ;
//   elems 0-7  = K kk + h*8 .. +7 ; elems 8-15 = K kk+16 + h*8 .. +7
// C/D: elem r -> M = h*8 + r, N = l  (per column tile)
// ---------------------------------------------------------------------------
template <int NT>
__global__ void k_gemm_wmma_f16(const float* __restrict__ A, int lda,
                                const _Float16* __restrict__ Bp,
                                float* __restrict__ C, int ldc, int K) {
    const int mtile = blockIdx.x;
    const int lane  = threadIdx.x;            // 0..31 (wave32)
    const int h     = lane >> 4;
    const int l     = lane & 15;
    const int row   = mtile * 16 + l;
    const int kc    = (K + 31) >> 5;

    v8f acc[NT];
#pragma unroll
    for (int t = 0; t < NT; ++t) acc[t] = (v8f){};

    const float* __restrict__ arow = A + (size_t)row * lda;
    const int kfull = K >> 5;

    for (int kchunk = 0; kchunk < kfull; ++kchunk) {
        const int kk = kchunk << 5;
        // A fragment: 4 x float4 (16B-aligned, fully in-bounds)
        const float4 p0 = *(const float4*)(arow + kk + h * 8);
        const float4 p1 = *(const float4*)(arow + kk + h * 8 + 4);
        const float4 p2 = *(const float4*)(arow + kk + 16 + h * 8);
        const float4 p3 = *(const float4*)(arow + kk + 16 + h * 8 + 4);
        v16h a;
        a[0] = (_Float16)p0.x; a[1] = (_Float16)p0.y; a[2]  = (_Float16)p0.z; a[3]  = (_Float16)p0.w;
        a[4] = (_Float16)p1.x; a[5] = (_Float16)p1.y; a[6]  = (_Float16)p1.z; a[7]  = (_Float16)p1.w;
        a[8] = (_Float16)p2.x; a[9] = (_Float16)p2.y; a[10] = (_Float16)p2.z; a[11] = (_Float16)p2.w;
        a[12] = (_Float16)p3.x; a[13] = (_Float16)p3.y; a[14] = (_Float16)p3.z; a[15] = (_Float16)p3.w;
#pragma unroll
        for (int t = 0; t < NT; ++t) {
            v16h b = *(const v16h*)(Bp + (((size_t)t * kc + kchunk) * 32 + lane) * 16);
            acc[t] = __builtin_amdgcn_wmma_f32_16x16x32_f16(false, a, false, b,
                                                            (short)0, acc[t], false, false);
        }
    }

    if (K & 31) {   // tail chunk: branchless clamped loads (no EXEC masking)
        const int kk = kfull << 5;
        v16h a;
#pragma unroll
        for (int i = 0; i < 8; ++i) {
            int k0 = kk + h * 8 + i;
            int k1 = kk + 16 + h * 8 + i;
            float f0 = arow[k0 < K ? k0 : 0];
            float f1 = arow[k1 < K ? k1 : 0];
            if (k0 >= K) f0 = 0.f;
            if (k1 >= K) f1 = 0.f;
            a[i]     = (_Float16)f0;
            a[8 + i] = (_Float16)f1;
        }
#pragma unroll
        for (int t = 0; t < NT; ++t) {
            v16h b = *(const v16h*)(Bp + (((size_t)t * kc + kfull) * 32 + lane) * 16);
            acc[t] = __builtin_amdgcn_wmma_f32_16x16x32_f16(false, a, false, b,
                                                            (short)0, acc[t], false, false);
        }
    }

    const int m0 = mtile * 16 + h * 8;
#pragma unroll
    for (int t = 0; t < NT; ++t) {
        const int col = t * 16 + l;
#pragma unroll
        for (int r = 0; r < 8; ++r)
            C[(size_t)(m0 + r) * ldc + col] = acc[t][r];
    }
}

// ---------------------------------------------------------------------------
// Edge-parallel scatter-add: acc[d] += norm[e] * h[s]   (4 feats / thread)
// ---------------------------------------------------------------------------
__global__ void k_scatter(const int* __restrict__ src, const int* __restrict__ dst,
                          const float* __restrict__ nrm, const float* __restrict__ h,
                          float* __restrict__ acc, int Fdim) {
    long long tid = (long long)blockIdx.x * blockDim.x + threadIdx.x;
    const int per = Fdim >> 2;
    const long long total = (long long)EE * per;
    if (tid >= total) return;
    const int e = (int)(tid / per);
    const int f = (int)(tid % per) << 2;
    const float w = nrm[e];
    const int s = src[e], d = dst[e];
    const float4 v = *(const float4*)(h + (size_t)s * Fdim + f);
    float* o = acc + (size_t)d * Fdim + f;
    atomicAdd(o + 0, w * v.x);
    atomicAdd(o + 1, w * v.y);
    atomicAdd(o + 2, w * v.z);
    atomicAdd(o + 3, w * v.w);
}

// ---------------------------------------------------------------------------
// Finalize conv: out = [relu](acc + dinv[i]^2 * pre + bias[f])
// ---------------------------------------------------------------------------
__global__ void k_finalize(const float* __restrict__ acc, const float* __restrict__ pre,
                           const float* __restrict__ dinv, const float* __restrict__ bias,
                           float* __restrict__ out, int Fdim, int relu) {
    long long tid = (long long)blockIdx.x * blockDim.x + threadIdx.x;
    const long long total = (long long)NN * Fdim;
    if (tid >= total) return;
    const int i = (int)(tid / Fdim);
    const int f = (int)(tid % Fdim);
    const float sn = dinv[i] * dinv[i];
    float v = acc[tid] + sn * pre[tid] + bias[f];
    out[tid] = relu ? fmaxf(v, 0.f) : v;
}

// ---------------------------------------------------------------------------
// Heads pre-projection: attr_pre/att_pre[i,c] = feat[i,:] . W[:,c]  (C=3)
// ---------------------------------------------------------------------------
__global__ void k_heads_pre(const float* __restrict__ feat,
                            const float* __restrict__ Wr, const float* __restrict__ Wk,
                            float* __restrict__ attr_pre, float* __restrict__ att_pre) {
    int tid = blockIdx.x * blockDim.x + threadIdx.x;
    if (tid >= NN * CC) return;
    const int i = tid / CC, c = tid % CC;
    const float* fr = feat + (size_t)i * H2C;
    float sr = 0.f, sk = 0.f;
#pragma unroll 8
    for (int f = 0; f < H2C; ++f) {
        float x = fr[f];
        sr = fmaf(x, Wr[f * CC + c], sr);
        sk = fmaf(x, Wk[f * CC + c], sk);
    }
    attr_pre[tid] = sr;
    att_pre[tid]  = sk;
}

__global__ void k_heads_scatter(const int* __restrict__ src, const int* __restrict__ dst,
                                const float* __restrict__ nrm,
                                const float* __restrict__ attr_pre,
                                const float* __restrict__ att_pre,
                                float* __restrict__ attrAcc, float* __restrict__ attAcc) {
    int e = blockIdx.x * blockDim.x + threadIdx.x;
    if (e >= EE) return;
    const float w = nrm[e];
    const int s = src[e], d = dst[e];
#pragma unroll
    for (int c = 0; c < CC; ++c) {
        atomicAdd(&attrAcc[(size_t)d * CC + c], w * attr_pre[(size_t)s * CC + c]);
        atomicAdd(&attAcc [(size_t)d * CC + c], w * att_pre [(size_t)s * CC + c]);
    }
}

__global__ void k_heads_final(const float* __restrict__ attrAcc, const float* __restrict__ attAcc,
                              const float* __restrict__ attr_pre, const float* __restrict__ att_pre,
                              const float* __restrict__ dinv,
                              const float* __restrict__ b_attr, const float* __restrict__ b_attk,
                              float* __restrict__ out_attr, float* __restrict__ out_att) {
    int i = blockIdx.x * blockDim.x + threadIdx.x;
    if (i >= NN) return;
    const float sn = dinv[i] * dinv[i];
    float a[CC], t[CC];
#pragma unroll
    for (int c = 0; c < CC; ++c) {
        a[c] = attrAcc[(size_t)i * CC + c] + sn * attr_pre[(size_t)i * CC + c] + b_attr[c];
        t[c] = attAcc [(size_t)i * CC + c] + sn * att_pre [(size_t)i * CC + c] + b_attk[c];
    }
    float m = fmaxf(a[0], fmaxf(a[1], a[2]));
    float lse = logf(expf(a[0] - m) + expf(a[1] - m) + expf(a[2] - m));
#pragma unroll
    for (int c = 0; c < CC; ++c) {
        out_attr[(size_t)i * CC + c] = a[c] - m - lse;
        out_att [(size_t)i * CC + c] = t[c];
    }
}

// ---------------------------------------------------------------------------
// Edge scores: res[e] = dot(feat[tot1[e]], feat[tot0[e]])  (64-dim)
// ---------------------------------------------------------------------------
__global__ void k_edge_scores(const int* __restrict__ pos, const int* __restrict__ neg,
                              const float* __restrict__ feat, float* __restrict__ res) {
    int e = blockIdx.x * blockDim.x + threadIdx.x;
    if (e >= 2 * PEE) return;
    int sj, si;
    if (e < PEE) { sj = pos[e];        si = pos[PEE + e]; }
    else         { int q = e - PEE; sj = neg[q]; si = neg[PEE + q]; }
    const float4* A = (const float4*)(feat + (size_t)sj * H2C);
    const float4* B = (const float4*)(feat + (size_t)si * H2C);
    float s = 0.f;
#pragma unroll
    for (int q = 0; q < H2C / 4; ++q) {
        float4 x = A[q], y = B[q];
        s += x.x * y.x + x.y * y.y + x.z * y.z + x.w * y.w;
    }
    res[e] = s;
}

// ---------------------------------------------------------------------------
// Launch
// ---------------------------------------------------------------------------
extern "C" void kernel_launch(void* const* d_in, const int* in_sizes, int n_in,
                              void* d_out, int out_size, void* d_ws, size_t ws_size,
                              hipStream_t stream) {
    const float* x      = (const float*)d_in[0];
    const float* glove  = (const float*)d_in[1];
    const float* W1     = (const float*)d_in[2];
    const float* b1     = (const float*)d_in[3];
    const float* W2     = (const float*)d_in[4];
    const float* b2     = (const float*)d_in[5];
    const float* W_attr = (const float*)d_in[6];
    const float* b_attr = (const float*)d_in[7];
    const float* W_attk = (const float*)d_in[8];
    const float* b_attk = (const float*)d_in[9];
    const int*   edge   = (const int*)d_in[10];   // [2,E]
    const int*   pos    = (const int*)d_in[11];   // [2,PE]
    const int*   neg    = (const int*)d_in[12];   // [2,PE]
    const int* e_src = edge;
    const int* e_dst = edge + EE;

    // Output layout: res[800000] | log_softmax(attr)[150000] | att[150000] | feat[3200000]
    float* out      = (float*)d_out;
    float* out_res  = out;
    float* out_attr = out + (size_t)2 * PEE;
    float* out_att  = out + (size_t)2 * PEE + (size_t)NN * CC;
    float* feat     = out + (size_t)2 * PEE + (size_t)2 * NN * CC;

    // Workspace layout (float-equivalents; all offsets 16-float = 64B aligned)
    float* ws = (float*)d_ws;
    float*    Weff     = ws;                      //    64,000 f32
    _Float16* Bp1      = (_Float16*)(ws + 64000); //    65,536 f16 (= 32,768 f32)
    _Float16* Bp2      = (_Float16*)(ws + 96768); //     8,192 f16 (=  4,096 f32)
    float*    deg      = ws + 100864;             //    50,000
    float*    dinv     = deg      + 50000;        //    50,000
    float*    nrm      = dinv     + 50000;        // 1,600,000
    float*    h1pre    = nrm      + 1600000;      // 6,400,000
    float*    acc1     = h1pre    + 6400000;      // 6,400,000 (becomes h1 in-place)
    float*    h2pre    = acc1     + 6400000;      // 3,200,000
    float*    acc2     = h2pre    + 3200000;      // 3,200,000
    float*    attr_pre = acc2     + 3200000;      //   150,000
    float*    att_pre  = attr_pre + 150000;       //   150,000
    float*    attrAcc  = att_pre  + 150000;       //   150,000
    float*    attAcc   = attrAcc  + 150000;       //   150,000

    const int B = 256;

    // ---- init accumulators (deterministic each call) ----
    k_fill<<<CDIV(NN, B), B, 0, stream>>>(deg, NN, 1.0f);                        // self-loop
    k_fill<<<CDIV((long long)NN * H1C, B), B, 0, stream>>>(acc1, (long long)NN * H1C, 0.f);
    k_fill<<<CDIV((long long)NN * H2C, B), B, 0, stream>>>(acc2, (long long)NN * H2C, 0.f);
    k_fill<<<CDIV(NN * CC, B), B, 0, stream>>>(attrAcc, NN * CC, 0.f);
    k_fill<<<CDIV(NN * CC, B), B, 0, stream>>>(attAcc,  NN * CC, 0.f);

    // ---- fold glove into W1, pack B operands into WMMA fragments ----
    k_weff<<<CDIV(FF * H1C, B), B, 0, stream>>>(glove, W1, Weff);
    k_packB<<<CDIV(8 * 16 * 512, B), B, 0, stream>>>(Weff, H1C, FF,  H1C / 16, Bp1);
    k_packB<<<CDIV(4 *  4 * 512, B), B, 0, stream>>>(W2,   H2C, H1C, H2C / 16, Bp2);

    // ---- gcn_norm ----
    k_deg <<<CDIV(EE, B), B, 0, stream>>>(e_dst, deg);
    k_dinv<<<CDIV(NN, B), B, 0, stream>>>(deg, dinv);
    k_norm<<<CDIV(EE, B), B, 0, stream>>>(e_src, e_dst, dinv, nrm);

    // ---- layer 1: h1pre = x @ Weff  (WMMA, K=500 -> 15 full chunks + tail) ----
    k_gemm_wmma_f16<8><<<NN / 16, 32, 0, stream>>>(x, FF, Bp1, h1pre, H1C, FF);
    k_scatter<<<CDIV((long long)EE * (H1C / 4), B), B, 0, stream>>>(
        e_src, e_dst, nrm, h1pre, acc1, H1C);
    k_finalize<<<CDIV((long long)NN * H1C, B), B, 0, stream>>>(
        acc1, h1pre, dinv, b1, acc1, H1C, /*relu=*/1);   // h1 in-place in acc1

    // ---- layer 2: h2pre = h1 @ W2 (WMMA, K=128, no tail) ----
    k_gemm_wmma_f16<4><<<NN / 16, 32, 0, stream>>>(acc1, H1C, Bp2, h2pre, H2C, H1C);
    k_scatter<<<CDIV((long long)EE * (H2C / 4), B), B, 0, stream>>>(
        e_src, e_dst, nrm, h2pre, acc2, H2C);
    k_finalize<<<CDIV((long long)NN * H2C, B), B, 0, stream>>>(
        acc2, h2pre, dinv, b2, feat, H2C, /*relu=*/0);   // feat -> d_out

    // ---- heads (C=3) ----
    k_heads_pre<<<CDIV(NN * CC, B), B, 0, stream>>>(feat, W_attr, W_attk, attr_pre, att_pre);
    k_heads_scatter<<<CDIV(EE, B), B, 0, stream>>>(
        e_src, e_dst, nrm, attr_pre, att_pre, attrAcc, attAcc);
    k_heads_final<<<CDIV(NN, B), B, 0, stream>>>(
        attrAcc, attAcc, attr_pre, att_pre, dinv, b_attr, b_attk, out_attr, out_att);

    // ---- edge dot-product scores ----
    k_edge_scores<<<CDIV(2 * PEE, B), B, 0, stream>>>(pos, neg, feat, out_res);

    (void)in_sizes; (void)n_in; (void)out_size; (void)ws_size;
}
